// RelationalTransformer_23295902614245
// MI455X (gfx1250) — matrix operand order repaired and checked by reading, not compiled, not run on hardware
//
#include <hip/hip_runtime.h>
#include <hip/hip_bf16.h>
#include <cstdint>

// ---------------------------------------------------------------------------
// RelationalTransformer forward for MI455X (gfx1250), wave32 + WMMA f16.
//
// Precision strategy: all GEMMs run A(f16) x W(f16) with f32 accumulation via
// V_WMMA_F32_16X16X32_F16 (8192 MAC/inst, 8x the f32 16x16x4 form). Residual
// stream, norms, softmax and loss stay f32. Weights (~13MB) are converted to
// f16 once per launch; everything is L2-resident on a 192MB L2.
//
// Input order (harness): 12 top-level arrays in setup_inputs() order, then
// params flattened like jax tree_leaves (dict keys sorted alphabetically):
//   blocks[0..3]: per block 25 leaves:
//     col:{k_norm,norm,q_norm,wk,wo,wq,wv} feat:{...} ffn:{w1,w2,w3}
//     ffn_norm nbr:{...}
//   dec:{boolean(W,b),datetime,number,text}  -> idx 112..119
//   enc:{boolean,col_name,datetime,number,text}(W,b) -> 120..129
//   mask_embs:{boolean,datetime,number,text} -> 130..133
//   norm:{boolean,col_name,datetime,number,text} -> 134..138
//   norm_out -> 139
// ---------------------------------------------------------------------------

typedef _Float16 half_t;
typedef __attribute__((ext_vector_type(16))) _Float16 v16h;
typedef __attribute__((ext_vector_type(8)))  _Float16 v8h;
typedef __attribute__((ext_vector_type(8)))  float    v8f;

#define B_   4
#define S_   1024
#define K_   8
#define D_   256
#define H_   8
#define HD_  32
#define DFF_ 1024
#define DT_  384
#define M_   (B_ * S_)
#define EPSF 1.1920929e-07f
#define NEGF (-1e30f)

__device__ inline v16h cat8(v8h lo, v8h hi) {
  return __builtin_shufflevector(lo, hi, 0,1,2,3,4,5,6,7,8,9,10,11,12,13,14,15);
}

// blockDim must be 256 (8 waves). red[] is 8 floats of shared memory.
__device__ inline float blockReduceSum(float v, float* red) {
  #pragma unroll
  for (int off = 16; off; off >>= 1) v += __shfl_xor(v, off, 32);
  __syncthreads();                       // protect red[] from a previous call
  if ((threadIdx.x & 31) == 0) red[threadIdx.x >> 5] = v;
  __syncthreads();
  float tot = 0.f;
  #pragma unroll
  for (int i = 0; i < 8; ++i) tot += red[i];
  return tot;
}

// --------------------------- simple converters -----------------------------
__global__ void cvt_f32_f16_k(const float* __restrict__ a, half_t* __restrict__ o, int n) {
  int i = blockIdx.x * blockDim.x + threadIdx.x;
  if (i < n) o[i] = (half_t)a[i];
}

// --------------------------- mask bit precompute ---------------------------
__global__ void build_masks_k(const int* __restrict__ node, const int* __restrict__ f2p,
                              const int* __restrict__ colI, const int* __restrict__ tabI,
                              const unsigned char* __restrict__ pad,
                              unsigned* __restrict__ mF, unsigned* __restrict__ mN,
                              unsigned* __restrict__ mC) {
  const int words = S_ >> 5;
  int gid = blockIdx.x * blockDim.x + threadIdx.x;
  if (gid >= B_ * S_ * words) return;
  int kw = gid % words;
  int q  = (gid / words) % S_;
  int b  = gid / (words * S_);
  int nq = node[b * S_ + q];
  int cq = colI[b * S_ + q], tq = tabI[b * S_ + q];
  bool pq = pad[b * S_ + q] != 0;
  int fq[K_];
  #pragma unroll
  for (int j = 0; j < K_; ++j) fq[j] = f2p[(size_t)(b * S_ + q) * K_ + j];
  unsigned wF = 0u, wN = 0u, wC = 0u;
  for (int kk = 0; kk < 32; ++kk) {
    int k = kw * 32 + kk;
    int nk = node[b * S_ + k];
    bool pk = pad[b * S_ + k] != 0;
    bool pd = (!pq) && (!pk);
    bool same_node = (nq == nk);
    bool kvin = false, qin = false;
    #pragma unroll
    for (int j = 0; j < K_; ++j) {
      kvin = kvin || (nk == fq[j]);
      qin  = qin  || (nq == f2p[(size_t)(b * S_ + k) * K_ + j]);
    }
    bool sct = (cq == colI[b * S_ + k]) && (tq == tabI[b * S_ + k]);
    if ((same_node || kvin) && pd) wF |= (1u << kk);
    if (qin && pd)                 wN |= (1u << kk);
    if (sct && pd)                 wC |= (1u << kk);
  }
  mF[gid] = wF; mN[gid] = wN; mC[gid] = wC;
}

// ------------------------------ WMMA GEMM ----------------------------------
// C[M,N] = A[M,K](f16) * W[N,K]^T(f16) (+bias[N]) (+residual[M,N])
// Optional outputs: f32, f16 row-major, and f16 transposed-per-batch
// (CT16[(b*N + n)*S_ + s], row = b*S_ + s) used for V^T.
// One wave computes a 16x64 strip: 4 WMMA accumulators reuse the A fragment.
__global__ __launch_bounds__(256) void gemm_wmma_k(
    const half_t* __restrict__ A, const half_t* __restrict__ W,
    const float* __restrict__ bias, const float* __restrict__ residual,
    float* __restrict__ C32, half_t* __restrict__ C16, half_t* __restrict__ CT16,
    int M, int N, int K) {
  int wid  = (int)((blockIdx.x * blockDim.x + threadIdx.x) >> 5);
  int lane = threadIdx.x & 31;
  int ng = N >> 6;
  int tm = wid / ng, tg = wid % ng;
  if (tm >= (M >> 4)) return;                      // wave-uniform
  int l15 = lane & 15, lh = lane >> 4;
  v8f acc[4] = {};
  const half_t* Ar = A + (size_t)(tm * 16 + l15) * K;
  for (int kb = 0; kb < K; kb += 32) {
    // Unconditional speculative prefetch of the next K-step: TH=0 prefetches
    // are dropped silently on bad addresses (ISA 10.5), so no guard branch —
    // this keeps the loop a single basic block for the scheduler.
    __builtin_prefetch(Ar + kb + 32 + lh * 8, 0, 1);
    __builtin_prefetch(W + (size_t)(tg * 64 + l15) * K + kb + 32 + lh * 16, 0, 1);
    // A fragment (16x32 f16): lane row = l15; elems 0..7 -> k=lh*8+e,
    // elems 8..15 -> k=16+lh*8+(e-8). Two contiguous 16B loads.
    v16h af = cat8(*(const v8h*)(Ar + kb + lh * 8),
                   *(const v8h*)(Ar + kb + 16 + lh * 8));
    #pragma unroll
    for (int t = 0; t < 4; ++t) {
      // B fragment (32x16 f16): lane col n = l15; k same as A fragment.
      const half_t* Wp = W + (size_t)(tg * 64 + t * 16 + l15) * K + kb;
      v16h bf = cat8(*(const v8h*)(Wp + lh * 8), *(const v8h*)(Wp + 16 + lh * 8));
      acc[t] = __builtin_amdgcn_wmma_f32_16x16x32_f16(
          false, af, false, bf, (short)0, acc[t], false, false);
    }
  }
  #pragma unroll
  for (int t = 0; t < 4; ++t) {
    int n = tg * 64 + t * 16 + l15;
    float bn = bias ? bias[n] : 0.f;
    #pragma unroll
    for (int v = 0; v < 8; ++v) {
      int r = tm * 16 + lh * 8 + v;                // C layout: row = v + 8*lh
      size_t idx = (size_t)r * N + n;
      float val = acc[t][v] + bn;
      if (residual) val += residual[idx];
      if (C32) C32[idx] = val;
      if (C16) C16[idx] = (half_t)val;
      if (CT16) {
        int bb = r >> 10, ss = r & (S_ - 1);
        CT16[((size_t)bb * N + n) * S_ + ss] = (half_t)val;
      }
    }
  }
}

// ------------------------------- RMSNorm -----------------------------------
__global__ __launch_bounds__(256) void rmsnorm_k(const float* __restrict__ x,
                                                 const float* __restrict__ w,
                                                 half_t* __restrict__ o16,
                                                 float* __restrict__ o32) {
  __shared__ float red[8];
  int row = blockIdx.x, tid = threadIdx.x;
  size_t idx = (size_t)row * D_ + tid;
  float v = x[idx];
  float tot = blockReduceSum(v * v, red);
  float o = v * rsqrtf(tot * (1.f / D_) + EPSF) * w[tid];
  o16[idx] = (half_t)o;
  if (o32) o32[idx] = o;
}

// Per-head RMSNorm of q/k (HD=32 == one wave per head) + V transpose to f16.
__global__ __launch_bounds__(256) void qkvnorm_k(
    const float* __restrict__ q32, const float* __restrict__ k32,
    const float* __restrict__ v32, const float* __restrict__ qn,
    const float* __restrict__ kn, half_t* __restrict__ q16,
    half_t* __restrict__ k16, half_t* __restrict__ vT16) {
  int row = blockIdx.x, tid = threadIdx.x;
  int j = tid & 31;
  size_t idx = (size_t)row * D_ + tid;
  float qv = q32[idx], kv = k32[idx], vv = v32[idx];
  float sq = qv * qv, sk = kv * kv;
  #pragma unroll
  for (int off = 16; off; off >>= 1) {
    sq += __shfl_xor(sq, off, 32);
    sk += __shfl_xor(sk, off, 32);
  }
  q16[idx] = (half_t)(qv * rsqrtf(sq * (1.f / HD_) + EPSF) * qn[j]);
  k16[idx] = (half_t)(kv * rsqrtf(sk * (1.f / HD_) + EPSF) * kn[j]);
  int b = row >> 10, s = row & (S_ - 1);
  vT16[((size_t)b * D_ + tid) * S_ + s] = (half_t)vv;
}

// --------------------------- flash attention -------------------------------
// One wave per (b, h, 16-query tile). Per 32 keys: 2 score WMMAs, masked
// online softmax (half-wave shfl reductions), LDS transpose of P into an
// A-fragment (single-wave, fenced by s_wait_dscnt), 2 P*V WMMAs.
__global__ __launch_bounds__(128) void attn_wmma_k(
    const half_t* __restrict__ q16, const half_t* __restrict__ k16,
    const half_t* __restrict__ vT16, const unsigned* __restrict__ maskbits,
    half_t* __restrict__ out16) {
  __shared__ half_t Pbuf[4][32 * 16];
  int wloc = threadIdx.x >> 5;
  int wid  = blockIdx.x * (blockDim.x >> 5) + wloc;
  int lane = threadIdx.x & 31;
  const int qtiles = S_ >> 4;
  int qt = wid % qtiles;
  int h  = (wid / qtiles) % H_;
  int b  = wid / (qtiles * H_);
  if (b >= B_) return;                              // wave-uniform
  int l15 = lane & 15, lh = lane >> 4;
  const float scale = 0.17677669529663687f;         // 1/sqrt(HD)

  const half_t* qp = q16 + ((size_t)(b * S_) + qt * 16 + l15) * D_ + h * 32;
  v16h afq = cat8(*(const v8h*)(qp + lh * 8), *(const v8h*)(qp + 16 + lh * 8));

  v8f o0 = {}, o1 = {};
  float mrow[8], lrow[8];
  #pragma unroll
  for (int v = 0; v < 8; ++v) { mrow[v] = NEGF; lrow[v] = 0.f; }

  const int mwords = S_ >> 5;
  const unsigned* mrowp = maskbits + ((size_t)(b * S_) + qt * 16 + l15) * mwords;
  half_t* P = &Pbuf[wloc][0];

  for (int kt = 0; kt < mwords; ++kt) {             // 32 keys / iteration
    // Unconditional speculative prefetch of next iteration's K rows / V^T
    // lines (safe past the end: speculative prefetches drop on translate).
    __builtin_prefetch(
        k16 + ((size_t)(b * S_) + (kt + 1) * 32 + l15) * D_ + h * 32 + lh * 16, 0, 1);
    __builtin_prefetch(
        vT16 + ((size_t)(b * D_) + h * 32 + l15) * S_ + (kt + 1) * 32 + lh * 16, 0, 1);
    v8f sc[2];
    #pragma unroll
    for (int t = 0; t < 2; ++t) {
      const half_t* kp = k16 + ((size_t)(b * S_) + kt * 32 + t * 16 + l15) * D_
                              + h * 32;
      v16h bk = cat8(*(const v8h*)(kp + lh * 8), *(const v8h*)(kp + 16 + lh * 8));
      v8f z = {};
      sc[t] = __builtin_amdgcn_wmma_f32_16x16x32_f16(
          false, afq, false, bk, (short)0, z, false, false);
    }
    unsigned mw = mrowp[kt];                        // lanes 0..15 hold rows 0..15
    float p0[8], p1[8];
    #pragma unroll
    for (int v = 0; v < 8; ++v) {
      int r = lh * 8 + v;
      unsigned w = __shfl(mw, r, 32);
      bool b0 = (w >> l15) & 1u;
      bool b1 = (w >> (16 + l15)) & 1u;
      float s0 = b0 ? sc[0][v] * scale : NEGF;
      float s1 = b1 ? sc[1][v] * scale : NEGF;
      float mx = fmaxf(s0, s1);
      #pragma unroll
      for (int off = 1; off < 16; off <<= 1) mx = fmaxf(mx, __shfl_xor(mx, off, 32));
      float mnew = fmaxf(mrow[v], mx);
      float corr = __expf(mrow[v] - mnew);
      float e0 = b0 ? __expf(s0 - mnew) : 0.f;
      float e1 = b1 ? __expf(s1 - mnew) : 0.f;
      float rs = e0 + e1;
      #pragma unroll
      for (int off = 1; off < 16; off <<= 1) rs += __shfl_xor(rs, off, 32);
      lrow[v] = lrow[v] * corr + rs;
      mrow[v] = mnew;
      o0[v] *= corr;
      o1[v] *= corr;
      p0[v] = e0; p1[v] = e1;
    }
    // Store P^T col-major [32 cols][16 rows]: one 16B store per tile per lane.
    v8h ph0, ph1;
    #pragma unroll
    for (int v = 0; v < 8; ++v) { ph0[v] = (half_t)p0[v]; ph1[v] = (half_t)p1[v]; }
    *(v8h*)(P + (l15)      * 16 + lh * 8) = ph0;
    *(v8h*)(P + (16 + l15) * 16 + lh * 8) = ph1;
    asm volatile("s_wait_dscnt 0" ::: "memory");    // single-wave LDS fence
    // Rebuild P as a 16x32 A-fragment: row = l15, cols per A layout.
    v16h ap;
    #pragma unroll
    for (int e = 0; e < 8; ++e) ap[e]     = P[(lh * 8 + e) * 16 + l15];
    #pragma unroll
    for (int e = 0; e < 8; ++e) ap[8 + e] = P[(16 + lh * 8 + e) * 16 + l15];
    #pragma unroll
    for (int t = 0; t < 2; ++t) {
      const half_t* vp = vT16 + ((size_t)(b * D_) + h * 32 + t * 16 + l15) * S_
                               + kt * 32;
      v16h bv = cat8(*(const v8h*)(vp + lh * 8), *(const v8h*)(vp + 16 + lh * 8));
      if (t == 0) o0 = __builtin_amdgcn_wmma_f32_16x16x32_f16(
                      false, ap, false, bv, (short)0, o0, false, false);
      else        o1 = __builtin_amdgcn_wmma_f32_16x16x32_f16(
                      false, ap, false, bv, (short)0, o1, false, false);
    }
  }
  #pragma unroll
  for (int v = 0; v < 8; ++v) {
    int r = qt * 16 + lh * 8 + v;
    float inv = (mrow[v] > -1e29f && lrow[v] > 0.f) ? 1.f / lrow[v] : 0.f;
    half_t* op = out16 + ((size_t)(b * S_) + r) * D_ + h * 32;
    op[l15]      = (half_t)(o0[v] * inv);
    op[16 + l15] = (half_t)(o1[v] * inv);
  }
}

// ------------------------------ SiLU gate ----------------------------------
__global__ void silu_mul_k(const float* __restrict__ g1, const float* __restrict__ g3,
                           half_t* __restrict__ h, int n) {
  int i = blockIdx.x * blockDim.x + threadIdx.x;
  if (i < n) {
    float a = g1[i];
    float s = a / (1.f + __expf(-a));
    h[i] = (half_t)(s * g3[i]);
  }
}

// ------------------------------- encoder -----------------------------------
__global__ __launch_bounds__(256) void encoder_k(
    const float* __restrict__ ecol, const float* __restrict__ etext,
    const float* __restrict__ numv, const float* __restrict__ dtv,
    const float* __restrict__ boolv, const int* __restrict__ sem,
    const unsigned char* __restrict__ msk, const unsigned char* __restrict__ pad,
    const float* __restrict__ ncol,
    const float* __restrict__ Wn, const float* __restrict__ bn,
    const float* __restrict__ nn, const float* __restrict__ en,
    const float* __restrict__ nt, const float* __restrict__ et,
    const float* __restrict__ Wd, const float* __restrict__ bd,
    const float* __restrict__ nd, const float* __restrict__ ed,
    const float* __restrict__ Wb, const float* __restrict__ bbias,
    const float* __restrict__ nb, const float* __restrict__ eb,
    float* __restrict__ x) {
  __shared__ float red[8];
  int row = blockIdx.x, tid = threadIdx.x;
  size_t idx = (size_t)row * D_ + tid;
  float c = ecol[idx];
  float tot = blockReduceSum(c * c, red);
  int st = sem[row];
  bool mk = msk[row] != 0, pd = pad[row] != 0;
  float xv = c * rsqrtf(tot * (1.f / D_) + EPSF) * ncol[tid] * (pd ? 0.f : 1.f);
  float ev;
  if (st == 1) {
    ev = etext[idx];
  } else {
    float val = (st == 0) ? numv[row] : (st == 2) ? dtv[row] : boolv[row];
    const float* W = (st == 0) ? Wn : (st == 2) ? Wd : Wb;
    const float* bi = (st == 0) ? bn : (st == 2) ? bd : bbias;
    ev = val * W[tid] + bi[tid];
  }
  float tot2 = blockReduceSum(ev * ev, red);
  const float* nw = (st == 0) ? nn : (st == 1) ? nt : (st == 2) ? nd : nb;
  const float* em = (st == 0) ? en : (st == 1) ? et : (st == 2) ? ed : eb;
  float encm = (!mk && !pd) ? 1.f : 0.f;
  float mskm = (mk && !pd) ? 1.f : 0.f;
  xv += ev * rsqrtf(tot2 * (1.f / D_) + EPSF) * nw[tid] * encm;
  xv += em[tid] * mskm;
  x[idx] = xv;
}

// ---------------------- scalar decoders + loss -----------------------------
__global__ __launch_bounds__(256) void dec_scalar_k(
    const float* __restrict__ xn,
    const float* __restrict__ Wnum, const float* __restrict__ bnum,
    const float* __restrict__ Wdt, const float* __restrict__ bdt,
    const float* __restrict__ Wbo, const float* __restrict__ bbo,
    const float* __restrict__ numv, const float* __restrict__ dtv,
    const float* __restrict__ boolv, const int* __restrict__ sem,
    const unsigned char* __restrict__ msk,
    float* __restrict__ onum, float* __restrict__ odt, float* __restrict__ obool,
    float* __restrict__ lacc, float* __restrict__ lcnt) {
  __shared__ float red[8];
  int row = blockIdx.x, tid = threadIdx.x;
  float v = xn[(size_t)row * D_ + tid];
  float dn = blockReduceSum(v * Wnum[tid], red);
  float dd = blockReduceSum(v * Wdt[tid], red);
  float db = blockReduceSum(v * Wbo[tid], red);
  if (tid == 0) {
    float yn = dn + bnum[0], yd = dd + bdt[0], yb = db + bbo[0];
    onum[row] = yn; odt[row] = yd; obool[row] = yb;
    int b = row >> 10;
    if (msk[row]) {
      atomicAdd(&lcnt[b], 1.f);
      int st = sem[row];
      if (st == 0) {
        float d = yn - numv[row], a = fabsf(d);
        atomicAdd(&lacc[b], (a < 1.f) ? 0.5f * d * d : a - 0.5f);
      } else if (st == 2) {
        float d = yd - dtv[row], a = fabsf(d);
        atomicAdd(&lacc[b], (a < 1.f) ? 0.5f * d * d : a - 0.5f);
      } else if (st == 3) {
        float y01 = (boolv[row] > 0.f) ? 1.f : 0.f;
        atomicAdd(&lacc[b], fmaxf(yb, 0.f) - yb * y01 + log1pf(__expf(-fabsf(yb))));
      }
    }
  }
}

__global__ void finalize_loss_k(const float* __restrict__ lacc,
                                const float* __restrict__ lcnt,
                                float* __restrict__ out) {
  if (threadIdx.x == 0 && blockIdx.x == 0) {
    float s = 0.f;
    for (int b = 0; b < B_; ++b) s += lacc[b] / fmaxf(lcnt[b], 1.f);
    out[0] = s / (float)B_;
  }
}

// ---------------------------------------------------------------------------
extern "C" void kernel_launch(void* const* d_in, const int* in_sizes, int n_in,
                              void* d_out, int out_size, void* d_ws, size_t ws_size,
                              hipStream_t stream) {
  (void)in_sizes; (void)n_in; (void)out_size; (void)ws_size;
  const int* node  = (const int*)d_in[0];
  const int* f2p   = (const int*)d_in[1];
  const int* colI  = (const int*)d_in[2];
  const int* tabI  = (const int*)d_in[3];
  const unsigned char* padB = (const unsigned char*)d_in[4];
  const int* sem   = (const int*)d_in[5];
  const unsigned char* mskB = (const unsigned char*)d_in[6];
  const float* colv  = (const float*)d_in[7];
  const float* numv  = (const float*)d_in[8];
  const float* txtv  = (const float*)d_in[9];
  const float* dtv   = (const float*)d_in[10];
  const float* boolv = (const float*)d_in[11];
  auto pf = [&](int i) -> const float* { return (const float*)d_in[i]; };

  // ---- workspace carve-up (deterministic every call) ----
  size_t wsoff = 0;
  auto alloc = [&](size_t bytes) -> void* {
    void* p = (void*)((char*)d_ws + wsoff);
    wsoff = (wsoff + bytes + 255) & ~(size_t)255;
    return p;
  };
  float*  x     = (float*)alloc((size_t)M_ * D_ * 4);
  half_t* xn16  = (half_t*)alloc((size_t)M_ * D_ * 2);
  float*  xn32  = (float*)alloc((size_t)M_ * D_ * 4);
  float*  q32   = (float*)alloc((size_t)M_ * D_ * 4);
  float*  k32   = (float*)alloc((size_t)M_ * D_ * 4);
  float*  v32   = (float*)alloc((size_t)M_ * D_ * 4);
  half_t* q16   = (half_t*)alloc((size_t)M_ * D_ * 2);
  half_t* k16   = (half_t*)alloc((size_t)M_ * D_ * 2);
  half_t* vT16  = (half_t*)alloc((size_t)M_ * D_ * 2);
  half_t* ao16  = (half_t*)alloc((size_t)M_ * D_ * 2);
  float*  g1    = (float*)alloc((size_t)M_ * DFF_ * 4);
  float*  g3    = (float*)alloc((size_t)M_ * DFF_ * 4);
  half_t* h16   = (half_t*)alloc((size_t)M_ * DFF_ * 2);
  float*  ecol  = (float*)alloc((size_t)M_ * D_ * 4);
  float*  etext = (float*)alloc((size_t)M_ * D_ * 4);
  half_t* cn16  = (half_t*)alloc((size_t)M_ * DT_ * 2);
  half_t* tx16  = (half_t*)alloc((size_t)M_ * DT_ * 2);
  unsigned* mF  = (unsigned*)alloc((size_t)B_ * S_ * (S_ / 32) * 4);
  unsigned* mN  = (unsigned*)alloc((size_t)B_ * S_ * (S_ / 32) * 4);
  unsigned* mC  = (unsigned*)alloc((size_t)B_ * S_ * (S_ / 32) * 4);
  float*  lossbuf = (float*)alloc(2 * B_ * 4);
  float*  lacc = lossbuf, *lcnt = lossbuf + B_;

  auto cvtW = [&](int idx, size_t elems) -> half_t* {
    half_t* dst = (half_t*)alloc(elems * sizeof(half_t));
    cvt_f32_f16_k<<<(int)((elems + 255) / 256), 256, 0, stream>>>(pf(idx), dst, (int)elems);
    return dst;
  };

  // ---- param indices (see header comment) ----
  const int PB = 12;
  const int layerOff[3] = {0, 7, 18};          // executed order: col, feat, nbr
  unsigned* maskOf[3];
  maskOf[0] = mC; maskOf[1] = mF; maskOf[2] = mN;

  // ---- convert all weights to f16 once ----
  half_t *wq16[4][3], *wk16[4][3], *wv16[4][3], *wo16[4][3];
  half_t *w1_16[4], *w2_16[4], *w3_16[4];
  for (int bb = 0; bb < 4; ++bb) {
    int base = PB + bb * 25;
    for (int li = 0; li < 3; ++li) {
      int lb = base + layerOff[li];
      wk16[bb][li] = cvtW(lb + 3, (size_t)D_ * D_);
      wo16[bb][li] = cvtW(lb + 4, (size_t)D_ * D_);
      wq16[bb][li] = cvtW(lb + 5, (size_t)D_ * D_);
      wv16[bb][li] = cvtW(lb + 6, (size_t)D_ * D_);
    }
    w1_16[bb] = cvtW(base + 14, (size_t)DFF_ * D_);
    w2_16[bb] = cvtW(base + 15, (size_t)D_ * DFF_);
    w3_16[bb] = cvtW(base + 16, (size_t)DFF_ * D_);
  }
  half_t* encColW16 = cvtW(122, (size_t)D_ * DT_);
  half_t* encTxtW16 = cvtW(128, (size_t)D_ * DT_);
  half_t* decTxtW16 = cvtW(118, (size_t)DT_ * D_);

  auto gemm = [&](const half_t* A, const half_t* W, const float* bias,
                  const float* residual, float* C32, half_t* C16, half_t* CT,
                  int M, int N, int K) {
    int waves = (M / 16) * (N / 64);
    gemm_wmma_k<<<waves / 8, 256, 0, stream>>>(A, W, bias, residual, C32, C16, CT, M, N, K);
  };

  // ---- output layout ----
  float* out    = (float*)d_out;
  float* o_loss = out;
  float* o_num  = out + 1;
  float* o_text = o_num + M_;
  float* o_dt   = o_text + (size_t)M_ * DT_;
  float* o_bool = o_dt + M_;

  hipMemsetAsync(lossbuf, 0, 2 * B_ * sizeof(float), stream);

  // ---- masks ----
  build_masks_k<<<(B_ * S_ * (S_ / 32) + 255) / 256, 256, 0, stream>>>(
      node, f2p, colI, tabI, padB, mF, mN, mC);

  // ---- encoders ----
  cvt_f32_f16_k<<<(M_ * DT_ + 255) / 256, 256, 0, stream>>>(colv, cn16, M_ * DT_);
  cvt_f32_f16_k<<<(M_ * DT_ + 255) / 256, 256, 0, stream>>>(txtv, tx16, M_ * DT_);
  gemm(cn16, encColW16, pf(123), nullptr, ecol, nullptr, nullptr, M_, D_, DT_);
  gemm(tx16, encTxtW16, pf(129), nullptr, etext, nullptr, nullptr, M_, D_, DT_);
  encoder_k<<<M_, 256, 0, stream>>>(
      ecol, etext, numv, dtv, boolv, sem, mskB, padB,
      pf(135),                            // norm.col_name
      pf(126), pf(127), pf(137), pf(132), // number: W,b,norm,emb
      pf(138), pf(133),                   // text: norm, emb
      pf(124), pf(125), pf(136), pf(131), // datetime
      pf(120), pf(121), pf(134), pf(130), // boolean
      x);

  // ---- transformer blocks ----
  for (int bb = 0; bb < 4; ++bb) {
    int base = PB + bb * 25;
    for (int li = 0; li < 3; ++li) {
      int lb = base + layerOff[li];
      rmsnorm_k<<<M_, 256, 0, stream>>>(x, pf(lb + 1), xn16, nullptr);
      gemm(xn16, wq16[bb][li], nullptr, nullptr, q32, nullptr, nullptr, M_, D_, D_);
      gemm(xn16, wk16[bb][li], nullptr, nullptr, k32, nullptr, nullptr, M_, D_, D_);
      gemm(xn16, wv16[bb][li], nullptr, nullptr, v32, nullptr, nullptr, M_, D_, D_);
      qkvnorm_k<<<M_, 256, 0, stream>>>(q32, k32, v32, pf(lb + 2), pf(lb + 0),
                                        q16, k16, vT16);
      attn_wmma_k<<<(B_ * H_ * (S_ / 16)) / 4, 128, 0, stream>>>(
          q16, k16, vT16, maskOf[li], ao16);
      gemm(ao16, wo16[bb][li], nullptr, x, x, nullptr, nullptr, M_, D_, D_);
    }
    rmsnorm_k<<<M_, 256, 0, stream>>>(x, pf(base + 17), xn16, nullptr);
    gemm(xn16, w1_16[bb], nullptr, nullptr, g1, nullptr, nullptr, M_, DFF_, D_);
    gemm(xn16, w3_16[bb], nullptr, nullptr, g3, nullptr, nullptr, M_, DFF_, D_);
    silu_mul_k<<<(M_ * DFF_ + 255) / 256, 256, 0, stream>>>(g1, g3, h16, M_ * DFF_);
    gemm(h16, w2_16[bb], nullptr, x, x, nullptr, nullptr, M_, D_, DFF_);
  }

  // ---- head ----
  rmsnorm_k<<<M_, 256, 0, stream>>>(x, pf(139), xn16, xn32);
  gemm(xn16, decTxtW16, pf(119), nullptr, o_text, nullptr, nullptr, M_, DT_, D_);
  dec_scalar_k<<<M_, 256, 0, stream>>>(
      xn32, pf(116), pf(117), pf(114), pf(115), pf(112), pf(113),
      numv, dtv, boolv, sem, mskB, o_num, o_dt, o_bool, lacc, lcnt);
  finalize_loss_k<<<1, 32, 0, stream>>>(lacc, lcnt, o_loss);
}
